// DecoderCBatchNorm_63788854280467
// MI455X (gfx1250) — matrix operand order repaired and checked by reading, not compiled, not run on hardware
//
#include <hip/hip_runtime.h>

typedef float    v8f  __attribute__((ext_vector_type(8)));
typedef float    v2f  __attribute__((ext_vector_type(2)));
typedef _Float16 v16h __attribute__((ext_vector_type(16)));
typedef _Float16 h8   __attribute__((ext_vector_type(8)));
typedef _Float16 h2   __attribute__((ext_vector_type(2)));

#define TPB 256
#define WPB 8
#define GRID_BLOCKS 256
#define TILES_TOTAL 4096   // B*T/16 = 16*4096/16

// h = relu(x) @ W[mat] + b[mat], two 16x16 WMMA (N tiles), K=32 f16.
// The K dimension is PERMUTED: staged position k holds original channel
// (k>>1) + 16*(k&1); the weight B-fragments are pre-swizzled with the same
// permutation, so each lane stages one packed f16x2 per row (ds_store_b32).
__device__ __forceinline__ void wave_gemm(const v8f& x0, const v8f& x1, int mat,
                                          _Float16* abuf, const _Float16* wlds,
                                          const float* blds, int ln, int n, int hb,
                                          v8f& o0, v8f& o1)
{
  // stage relu(x): lane's channels (n, n+16) -> adjacent positions (2n, 2n+1)
  #pragma unroll
  for (int r = 0; r < 8; ++r) {
    int m = r + 8 * hb;
    h2 pk;
    pk[0] = (_Float16)fmaxf(x0[r], 0.f);
    pk[1] = (_Float16)fmaxf(x1[r], 0.f);
    *(h2*)(abuf + m * 32 + n * 2) = pk;
  }
  // A fragment: lane<16 -> row=lane, K 0..15 ; lane>=16 -> K 16..31
  h8 lo = *(const h8*)(abuf + n * 32 + hb * 16);
  h8 hi = *(const h8*)(abuf + n * 32 + hb * 16 + 8);
  v16h a;
  #pragma unroll
  for (int i = 0; i < 8; ++i) { a[i] = lo[i]; a[i + 8] = hi[i]; }
  // B fragments: pre-swizzled contiguous 16 halves per lane
  const v16h bf0 = *(const v16h*)(wlds + ((mat * 2 + 0) * 32 + ln) * 16);
  const v16h bf1 = *(const v16h*)(wlds + ((mat * 2 + 1) * 32 + ln) * 16);
  float bb0 = blds[mat * 32 + n];
  float bb1 = blds[mat * 32 + n + 16];
  v8f c0, c1;
  #pragma unroll
  for (int i = 0; i < 8; ++i) { c0[i] = bb0; c1[i] = bb1; }
  o0 = __builtin_amdgcn_wmma_f32_16x16x32_f16(false, a, false, bf0, (short)0, c0, false, false);
  o1 = __builtin_amdgcn_wmma_f32_16x16x32_f16(false, a, false, bf1, (short)0, c1, false, false);
}

__global__ __launch_bounds__(TPB) void decoder_fused_kernel(
    const float* __restrict__ p, const float* __restrict__ c,
    const float* __restrict__ Cmat,
    const float* __restrict__ fcpW, const float* __restrict__ fcpB,
    const float* __restrict__ W0, const float* __restrict__ b0,
    const float* __restrict__ W1, const float* __restrict__ b1,
    const float* __restrict__ Wout, const float* __restrict__ bout,
    float* __restrict__ out)
{
  __shared__ _Float16 wlds[10 * 2 * 32 * 16];          // 20 KB: B-fragments, 10 matrices x 2 N tiles
  __shared__ float    blds[320];                       // biases for 10 matrices
  __shared__ __align__(16) unsigned char actraw[WPB][2048];  // per-wave activation / cf tile

  const int tid = threadIdx.x;

  // ---- Stage weights once per block: f32 -> f16 B-fragments with permuted K ----
  // B (32x16 f16) order used here: lanes 0-15 hold K=0..15 (2/VGPR), lanes 16-31 hold K=16..31; N = lane&15.
  // Position k sources original row (k>>1) + 16*(k&1)  =>  pair v sources rows (rp, rp+16).
  for (int i = tid; i < 5120; i += TPB) {
    int v   = i & 7;
    int lnw = (i >> 3) & 31;
    int tN  = (i >> 8) & 1;
    int mat = i >> 9;
    int nn  = (lnw & 15) + tN * 16;
    int rp  = (lnw >> 4) * 8 + v;   // source row pair base for positions (2v, 2v+1)
    const float* Ws = (mat < 5) ? (W0 + mat * 1024) : (W1 + (mat - 5) * 1024);
    wlds[((mat * 2 + tN) * 32 + lnw) * 16 + 2 * v]     = (_Float16)Ws[rp * 32 + nn];
    wlds[((mat * 2 + tN) * 32 + lnw) * 16 + 2 * v + 1] = (_Float16)Ws[(rp + 16) * 32 + nn];
  }
  for (int i = tid; i < 320; i += TPB) {
    int mat = i >> 5, ch = i & 31;
    blds[i] = (mat < 5) ? b0[mat * 32 + ch] : b1[(mat - 5) * 32 + ch];
  }
  __syncthreads();

  const int wid = tid >> 5;
  const int ln  = tid & 31;
  const int n   = ln & 15;   // channel within N tile / point index for gather phase
  const int hb  = ln >> 4;   // half select
  float*    cbuf = (float*)actraw[wid];
  _Float16* abuf = (_Float16*)actraw[wid];

  // ---- Loop-invariant per-lane constants ----
  // fc_p B fragment (4x16 f32): lane<16 holds K0,K1; lane>=16 holds K2,K3 (K=3 zero pad)
  v2f wp0, wp1;
  {
    int k0 = hb * 2;
    wp0[0] = (k0     < 3) ? fcpW[k0 * 32 + n]            : 0.f;
    wp0[1] = (k0 + 1 < 3) ? fcpW[(k0 + 1) * 32 + n]      : 0.f;
    wp1[0] = (k0     < 3) ? fcpW[k0 * 32 + n + 16]       : 0.f;
    wp1[1] = (k0 + 1 < 3) ? fcpW[(k0 + 1) * 32 + n + 16] : 0.f;
  }
  const float bp0 = fcpB[n], bp1 = fcpB[n + 16];
  const float wo0 = Wout[n], wo1 = Wout[n + 16];
  const float bo  = bout[0];

  const int waves = gridDim.x * WPB;
  for (int tile = blockIdx.x * WPB + wid; tile < TILES_TOTAL; tile += waves) {
    const int bIdx = tile >> 8;          // 256 tiles per batch
    const int t0   = (tile & 255) << 4;

    // ---- Gather phase: lane handles point = n, channels [hb*16, hb*16+16) ----
    const float* pp = p + ((bIdx * 4096) + t0 + n) * 3;
    const float px = pp[0], py = pp[1], pz = pp[2];
    const float psx = px * (1.f / 0.55f), psy = py * (1.f / 0.55f), psz = pz * (1.f / 0.55f);

    float cfl[16];
    #pragma unroll
    for (int i = 0; i < 16; ++i) cfl[i] = 0.f;

    #pragma unroll
    for (int l = 0; l < 4; ++l) {
      const float* Cm = Cmat + (bIdx * 4 + l) * 12;
      float pr0 = Cm[0] * psx + Cm[1] * psy + Cm[2] * psz;
      float pr1 = Cm[3] * psx + Cm[4] * psy + Cm[5] * psz;
      float inv = 1.f / (Cm[9] + 0.05f);
      float xg = (pr0 * inv + 1.f) * 63.5f;
      float yg = (pr1 * inv + 1.f) * 63.5f;
      xg = (xg >= 127.f) ? 126.9f : xg;  xg = (xg < 0.f) ? 0.f : xg;
      yg = (yg >= 127.f) ? 126.9f : yg;  yg = (yg < 0.f) ? 0.f : yg;
      float xl = rintf(xg - 0.5f), xr = rintf(xg + 0.5f);
      float yl = rintf(yg - 0.5f), yh = rintf(yg + 0.5f);
      int xil = (int)xl, xir = (int)xr, yil = (int)yl, yih = (int)yh;
      float dx = xr - xg, dy = yh - yg;
      const float* vbase = c + (size_t)(bIdx * 4 + l) * (128 * 128 * 32) + hb * 16;

      auto tap = [&](int xi, int yi, float w) {
        const float4* f = (const float4*)(vbase + ((size_t)xi * 128 + yi) * 32);
        #pragma unroll
        for (int j = 0; j < 4; ++j) {
          float4 q = f[j];
          cfl[4 * j + 0] += w * q.x; cfl[4 * j + 1] += w * q.y;
          cfl[4 * j + 2] += w * q.z; cfl[4 * j + 3] += w * q.w;
        }
      };
      tap(xil, yil, dx * dy);
      tap(xir, yil, (1.f - dx) * dy);
      tap(xil, yih, dx * (1.f - dy));
      tap(xir, yih, (1.f - dx) * (1.f - dy));
    }

    // stage cf tile [16][32] f32 to LDS, then pull into D-fragment layout (kept in regs, reused 5x)
    {
      float4* cd = (float4*)(cbuf + n * 32 + hb * 16);
      #pragma unroll
      for (int j = 0; j < 4; ++j) {
        float4 q; q.x = cfl[4*j]; q.y = cfl[4*j+1]; q.z = cfl[4*j+2]; q.w = cfl[4*j+3];
        cd[j] = q;
      }
    }
    v8f cf0, cf1;
    #pragma unroll
    for (int r = 0; r < 8; ++r) {
      int m = r + 8 * hb;
      cf0[r] = cbuf[m * 32 + n];
      cf1[r] = cbuf[m * 32 + n + 16];
    }

    // ---- fc_p via full-precision f32 WMMA (16x16x4): A = p tile padded to K=4 ----
    v2f pa;
    pa[0] = hb ? pz : px;
    pa[1] = hb ? 0.f : py;
    v8f c0s, c1s;
    #pragma unroll
    for (int i = 0; i < 8; ++i) { c0s[i] = bp0; c1s[i] = bp1; }
    v8f net0 = __builtin_amdgcn_wmma_f32_16x16x4_f32(false, pa, false, wp0, (short)0, c0s, false, false);
    v8f net1 = __builtin_amdgcn_wmma_f32_16x16x4_f32(false, pa, false, wp1, (short)0, c1s, false, false);
    net0 += cf0; net1 += cf1;

    // ---- 5 ResNet blocks: net += fc1(relu(fc0(relu(net)))) ; cf re-added between blocks ----
    #pragma unroll
    for (int blk = 0; blk < 5; ++blk) {
      v8f h0, h1, d0, d1;
      wave_gemm(net0, net1, blk,     abuf, wlds, blds, ln, n, hb, h0, h1);
      wave_gemm(h0,   h1,   5 + blk, abuf, wlds, blds, ln, n, hb, d0, d1);
      net0 += d0; net1 += d1;
      if (blk < 4) { net0 += cf0; net1 += cf1; }
    }

    // ---- fc_out: channel reduction across each 16-lane half (halves hold disjoint points) ----
    #pragma unroll
    for (int r = 0; r < 8; ++r) {
      float pr = fmaxf(net0[r], 0.f) * wo0 + fmaxf(net1[r], 0.f) * wo1;
      pr += __shfl_xor(pr, 1, 32);
      pr += __shfl_xor(pr, 2, 32);
      pr += __shfl_xor(pr, 4, 32);
      pr += __shfl_xor(pr, 8, 32);
      if (n == r) out[bIdx * 4096 + t0 + r + 8 * hb] = pr + bo;
    }
  }
}

extern "C" void kernel_launch(void* const* d_in, const int* in_sizes, int n_in,
                              void* d_out, int out_size, void* d_ws, size_t ws_size,
                              hipStream_t stream) {
  const float* p    = (const float*)d_in[0];
  // d_in[1] = z : unused by the reference
  const float* c    = (const float*)d_in[2];
  const float* Cm   = (const float*)d_in[3];
  const float* fpW  = (const float*)d_in[4];
  const float* fpB  = (const float*)d_in[5];
  const float* W0   = (const float*)d_in[6];
  const float* b0   = (const float*)d_in[7];
  const float* W1   = (const float*)d_in[8];
  const float* b1   = (const float*)d_in[9];
  const float* Wo   = (const float*)d_in[10];
  const float* bo   = (const float*)d_in[11];
  decoder_fused_kernel<<<GRID_BLOCKS, TPB, 0, stream>>>(
      p, c, Cm, fpW, fpB, W0, b0, W1, b1, Wo, bo, (float*)d_out);
}
